// Mamba_INR_50672024158848
// MI455X (gfx1250) — matrix operand order repaired and checked by reading, not compiled, not run on hardware
//
#include <hip/hip_runtime.h>
#include <math.h>

// ---------- types ----------
typedef _Float16 half_t;
typedef __attribute__((ext_vector_type(16))) _Float16 v16h;
typedef __attribute__((ext_vector_type(8)))  float    v8f;
typedef __attribute__((ext_vector_type(4)))  unsigned u32x4;
typedef __attribute__((ext_vector_type(4)))  float    f4_t;

union Frag { u32x4 q[2]; v16h v; };

#define FD 64
#define DSN 16
#define DI 128
#define DRN 4

// =====================================================================
// weight convert / pad  (f32 -> f16, zero-pad rows/cols to WMMA tiles)
// =====================================================================
__global__ void pad_convert_kernel(const float* __restrict__ src, half_t* __restrict__ dst,
                                   int srows, int scols, int drows, int dcols) {
  int idx = blockIdx.x * 256 + threadIdx.x;
  if (idx >= drows * dcols) return;
  int r = idx / dcols, c = idx % dcols;
  float v = (r < srows && c < scols) ? src[r * scols + c] : 0.0f;
  dst[idx] = (half_t)v;
}

__global__ void convert_f16_kernel(const float* __restrict__ src, half_t* __restrict__ dst, long n) {
  long idx = (long)blockIdx.x * 256 + threadIdx.x;
  if (idx < n) dst[idx] = (half_t)src[idx];
}

// =====================================================================
// 3x3 SAME conv + relu  (encoders; compute-light, VALU)
// =====================================================================
__global__ void conv3x3_relu_kernel(const float* __restrict__ in, const float* __restrict__ w,
                                    const float* __restrict__ bias, float* __restrict__ out,
                                    int Cin, int Hh, int Ww) {
  int idx = blockIdx.x * 256 + threadIdx.x;
  int total = 4 * 64 * Hh * Ww;
  if (idx >= total) return;
  int j  = idx % Ww;
  int i  = (idx / Ww) % Hh;
  int co = (idx / (Ww * Hh)) % 64;
  int b  = idx / (Ww * Hh * 64);
  float s = bias[co];
  for (int ci = 0; ci < Cin; ++ci) {
    const float* ip = in + ((size_t)(b * Cin + ci) * Hh) * Ww;
    const float* wp = w + ((size_t)(co * Cin + ci)) * 9;
    #pragma unroll
    for (int ky = 0; ky < 3; ++ky) {
      int y = i + ky - 1;
      if (y < 0 || y >= Hh) continue;
      #pragma unroll
      for (int kx = 0; kx < 3; ++kx) {
        int x = j + kx - 1;
        if (x < 0 || x >= Ww) continue;
        s += ip[y * Ww + x] * wp[ky * 3 + kx];
      }
    }
  }
  out[idx] = fmaxf(s, 0.0f);
}

// =====================================================================
// generic WMMA GEMM: C[M,N](f32) = A[M,K](f16, row-major) * W[N,K]^T (f16)
// one wave per 16-row strip, NT n-tiles of 16 cols each (compile-time)
// =====================================================================
template <int NT>
__global__ __launch_bounds__(128) void gemm_f16_kernel(const half_t* __restrict__ A,
                                                       const half_t* __restrict__ W,
                                                       float* __restrict__ C,
                                                       int M, int N, int K) {
  int lane = threadIdx.x & 31;
  int wv   = threadIdx.x >> 5;
  int mt   = blockIdx.x * 4 + wv;
  int n0   = blockIdx.y * 16 * NT;
  int arow = mt * 16 + (lane & 15);
  int kbA  = (lane >> 4) * 16;   // A frag: 16 contiguous halves
  int kbB  = (lane >> 4) * 16;   // B frag: 16 contiguous halves
  v8f acc[NT];
  #pragma unroll
  for (int i = 0; i < NT; ++i) acc[i] = (v8f)(0.0f);

  const half_t* arowp = A + (size_t)arow * K;
  for (int k0 = 0; k0 < K; k0 += 32) {
    Frag a;
    const u32x4* ap = (const u32x4*)(arowp + k0 + kbA);
    a.q[0] = ap[0];
    a.q[1] = ap[1];
    #pragma unroll
    for (int nt = 0; nt < NT; ++nt) {
      int ncol = n0 + nt * 16 + (lane & 15);
      Frag bf;
      const u32x4* bp = (const u32x4*)(W + (size_t)ncol * K + k0 + kbB);
      bf.q[0] = bp[0];
      bf.q[1] = bp[1];
      acc[nt] = __builtin_amdgcn_wmma_f32_16x16x32_f16(false, a.v, false, bf.v,
                                                       (short)0, acc[nt], false, false);
    }
  }
  int rbase = mt * 16 + (lane >> 4) * 8;
  #pragma unroll
  for (int nt = 0; nt < NT; ++nt) {
    int ncol = n0 + nt * 16 + (lane & 15);
    #pragma unroll
    for (int r = 0; r < 8; ++r)
      C[(size_t)(rbase + r) * N + ncol] = acc[nt][r];
  }
}

// =====================================================================
// depthwise causal conv (DC=4) + SiLU -> u (f16, GEMM2 input)
// =====================================================================
__global__ void dwconv_silu_kernel(const float* __restrict__ xz, const float* __restrict__ cw,
                                   const float* __restrict__ cb, half_t* __restrict__ uh, int L) {
  long idx = (long)blockIdx.x * 256 + threadIdx.x;
  long total = (long)4 * L * DI;
  if (idx >= total) return;
  int d   = (int)(idx & 127);
  long bt = idx >> 7;
  int t   = (int)(bt % L);
  float s = cb[d];
  #pragma unroll
  for (int k = 0; k < 4; ++k) {
    int tt = t - 3 + k;
    if (tt >= 0) s += xz[(bt - 3 + k) * 256 + d] * cw[d * 4 + k];
  }
  float u = s / (1.0f + __expf(-s));
  uh[idx] = (half_t)u;
}

// =====================================================================
// fused SSM scan: per-channel thread keeps 16 states + A row in VGPRs.
// fuses softplus(dt), depthwise-conv/SiLU recompute, and SiLU(z) gating;
// emits g (f16) = (y + u*D) * silu(z), the input of the output GEMM.
// =====================================================================
__global__ __launch_bounds__(128) void mamba_scan_kernel(
    const float* __restrict__ xz, const float* __restrict__ xdbl,
    const float* __restrict__ Alog, const float* __restrict__ Wdt,
    const float* __restrict__ bdt, const float* __restrict__ cw,
    const float* __restrict__ cb, const float* __restrict__ Dp,
    half_t* __restrict__ gh, int L) {
  int d = threadIdx.x;
  int b = blockIdx.x;

  float Av[DSN];
  #pragma unroll
  for (int n = 0; n < DSN; ++n) Av[n] = -__expf(Alog[d * DSN + n]);
  float wd0 = Wdt[d * 4], wd1 = Wdt[d * 4 + 1], wd2 = Wdt[d * 4 + 2], wd3 = Wdt[d * 4 + 3];
  float bd = bdt[d], Dd = Dp[d];
  float cw0 = cw[d * 4], cw1 = cw[d * 4 + 1], cw2 = cw[d * 4 + 2], cw3 = cw[d * 4 + 3];
  float cbd = cb[d];

  float h[DSN];
  #pragma unroll
  for (int n = 0; n < DSN; ++n) h[n] = 0.0f;
  float w0 = 0.0f, w1 = 0.0f, w2 = 0.0f;

  long base = (long)b * L;
  for (int t = 0; t < L; ++t) {
    long bt = base + t;
    const float* xrow = xz + bt * 256;
    __builtin_prefetch(xdbl + (bt + 8) * 48, 0, 3);
    __builtin_prefetch(xz + (bt + 8) * 256 + d, 0, 3);

    float xc = xrow[d];
    float z  = xrow[128 + d];
    const f4_t* xr4 = (const f4_t*)(xdbl + bt * 48);
    f4_t qd = xr4[0];
    float Bv[DSN], Cv[DSN];
    #pragma unroll
    for (int q = 0; q < 4; ++q) {
      f4_t tb = xr4[1 + q];
      f4_t tc = xr4[5 + q];
      #pragma unroll
      for (int e = 0; e < 4; ++e) { Bv[q * 4 + e] = tb[e]; Cv[q * 4 + e] = tc[e]; }
    }
    // dt = softplus(xdbl[:,0:4] @ Wdt[d] + bdt[d])
    float dl = bd + wd0 * qd[0] + wd1 * qd[1] + wd2 * qd[2] + wd3 * qd[3];
    float dt = (dl > 20.0f) ? dl : log1pf(__expf(dl));
    // causal depthwise conv + SiLU (rolling window)
    float cacc = cbd + cw0 * w0 + cw1 * w1 + cw2 * w2 + cw3 * xc;
    w0 = w1; w1 = w2; w2 = xc;
    float u = cacc / (1.0f + __expf(-cacc));
    float dtu = dt * u;
    float y = 0.0f;
    #pragma unroll
    for (int n = 0; n < DSN; ++n) {
      float e = __expf(dt * Av[n]);
      h[n] = e * h[n] + dtu * Bv[n];
      y += h[n] * Cv[n];
    }
    float g = (y + u * Dd) * (z / (1.0f + __expf(-z)));
    gh[bt * 128 + d] = (half_t)g;
  }
}

// =====================================================================
// fused INR kernel: 32 pixels x 4 shifts = 128-row MLP tile in LDS.
// gather -> L1(160->256,relu) -> L2(256->128,relu) -> L3(128->32)
// -> softmax over shifts -> blended output. WMMA from LDS fragments.
// =====================================================================
__device__ __forceinline__ void mlp_layer(const half_t* As, int lda,
                                          const half_t* __restrict__ Wg, int ldw,
                                          const float* __restrict__ bias,
                                          half_t* DstH, float* DstF, int ldd,
                                          int Mtiles, int Ntiles, int K, bool do_relu) {
  int lane = threadIdx.x & 31;
  int wv   = threadIdx.x >> 5;
  int kbA  = (lane >> 4) * 16;
  int kbB  = (lane >> 4) * 16;
  int total = Mtiles * Ntiles;
  for (int tile = wv; tile < total; tile += 4) {
    int mt = tile % Mtiles;
    int nt = tile / Mtiles;
    int arow = mt * 16 + (lane & 15);
    int ncol = nt * 16 + (lane & 15);
    v8f acc = (v8f)(0.0f);
    for (int k0 = 0; k0 < K; k0 += 32) {
      Frag a, bf;
      const u32x4* ap = (const u32x4*)(As + (size_t)arow * lda + k0 + kbA);
      a.q[0] = ap[0];
      a.q[1] = ap[1];
      const u32x4* bp = (const u32x4*)(Wg + (size_t)ncol * ldw + k0 + kbB);
      bf.q[0] = bp[0];
      bf.q[1] = bp[1];
      acc = __builtin_amdgcn_wmma_f32_16x16x32_f16(false, a.v, false, bf.v,
                                                   (short)0, acc, false, false);
    }
    float bv = bias[ncol];
    int rbase = mt * 16 + (lane >> 4) * 8;
    #pragma unroll
    for (int r = 0; r < 8; ++r) {
      float v = acc[r] + bv;
      if (do_relu) v = fmaxf(v, 0.0f);
      if (DstH) DstH[(size_t)(rbase + r) * ldd + ncol] = (half_t)v;
      else      DstF[(size_t)(rbase + r) * ldd + ncol] = v;
    }
  }
}

__global__ __launch_bounds__(128) void inr_mlp_kernel(
    const float* __restrict__ feat, const float* __restrict__ guide,
    const half_t* __restrict__ W1, const float* __restrict__ b1,
    const half_t* __restrict__ W2, const float* __restrict__ b2,
    const half_t* __restrict__ W3, const float* __restrict__ b3,
    float* __restrict__ out) {
  extern __shared__ char smem[];
  half_t* Xs  = (half_t*)smem;              // 128 x 160 f16 (40960 B)
  half_t* H1  = (half_t*)(smem + 40960);    // 128 x 256 f16 (65536 B)
  half_t* H2  = (half_t*)smem;              // 128 x 128 f16 (aliases Xs)
  float*  OUT = (float*)(smem + 40960);     // 128 x 32 f32 (aliases H1)

  // ---- Phase A: gather + build X (row = shift*32 + pixel_local) ----
  {
    int r  = threadIdx.x;           // 0..127
    int s  = r >> 5;                // shift 0..3
    int pl = r & 31;
    int p  = blockIdx.x * 32 + pl;  // global pixel over B*H*W
    int b  = p >> 14;
    int pix = p & 16383;
    int ii = pix >> 7;
    int jj = pix & 127;
    float cy = -1.0f + (2.0f * ii + 1.0f) / 128.0f;
    float cx = -1.0f + (2.0f * jj + 1.0f) / 128.0f;
    float vx = (s & 2) ? 1.0f : -1.0f;
    float vy = (s & 1) ? 1.0f : -1.0f;
    float cys = cy + vx * (1.0f / 32.0f);
    float cxs = cx + vy * (1.0f / 32.0f);
    float iyf = (cys + 1.0f) * 16.0f - 0.5f;   // h/2 = 16
    float ixf = (cxs + 1.0f) * 16.0f - 0.5f;
    int iyr = (int)rintf(iyf);                 // round half-to-even == jnp.round
    int ixr = (int)rintf(ixf);
    bool valid = (iyr >= 0 && iyr < 32 && ixr >= 0 && ixr < 32);
    int iyc = min(max(iyr, 0), 31);
    int ixc = min(max(ixr, 0), 31);

    half_t* xrow = Xs + (size_t)r * 160;
    const float* fsrc = feat + (size_t)b * 64 * 1024 + iyc * 32 + ixc;
    for (int c = 0; c < 64; ++c)
      xrow[c] = valid ? (half_t)fsrc[(size_t)c * 1024] : (half_t)0.0f;
    const float* gsrc = guide + (size_t)b * 64 * 16384 + ii * 128 + jj;
    for (int c = 0; c < 64; ++c)
      xrow[64 + c] = (half_t)gsrc[(size_t)c * 16384];
    float qc0 = valid ? (-1.0f + (2.0f * iyc + 1.0f) / 32.0f) : 0.0f;
    float qc1 = valid ? (-1.0f + (2.0f * ixc + 1.0f) / 32.0f) : 0.0f;
    xrow[128] = (half_t)((cy - qc0) * 32.0f);
    xrow[129] = (half_t)((cx - qc1) * 32.0f);
    #pragma unroll
    for (int c = 130; c < 160; ++c) xrow[c] = (half_t)0.0f;
  }
  __syncthreads();

  // ---- Layer 1: X(128x160) @ W1^T -> H1(128x256), relu ----
  mlp_layer(Xs, 160, W1, 160, b1, H1, nullptr, 256, 8, 16, 160, true);
  __syncthreads();
  // ---- Layer 2: H1 @ W2^T -> H2(128x128), relu ----
  mlp_layer(H1, 256, W2, 256, b2, H2, nullptr, 128, 8, 8, 256, true);
  __syncthreads();
  // ---- Layer 3: H2 @ W3^T -> OUT(128x32) f32 ----
  mlp_layer(H2, 128, W3, 128, b3, nullptr, OUT, 32, 8, 2, 128, false);
  __syncthreads();

  // ---- Phase C: softmax over 4 shifts (channel 31) + blend ----
  {
    int t  = threadIdx.x;
    int pl = t & 31;
    int c0 = t >> 5;
    float lw0 = OUT[(0 * 32 + pl) * 32 + 31];
    float lw1 = OUT[(1 * 32 + pl) * 32 + 31];
    float lw2 = OUT[(2 * 32 + pl) * 32 + 31];
    float lw3 = OUT[(3 * 32 + pl) * 32 + 31];
    float m  = fmaxf(fmaxf(lw0, lw1), fmaxf(lw2, lw3));
    float e0 = __expf(lw0 - m), e1 = __expf(lw1 - m), e2 = __expf(lw2 - m), e3 = __expf(lw3 - m);
    float inv = 1.0f / (e0 + e1 + e2 + e3);
    float wt0 = e0 * inv, wt1 = e1 * inv, wt2 = e2 * inv, wt3 = e3 * inv;

    int p  = blockIdx.x * 32 + pl;
    int b  = p >> 14;
    int pix = p & 16383;
    int ii = pix >> 7;
    int jj = pix & 127;
    for (int c = c0; c < 31; c += 4) {
      float acc = OUT[(0 * 32 + pl) * 32 + c] * wt0 +
                  OUT[(1 * 32 + pl) * 32 + c] * wt1 +
                  OUT[(2 * 32 + pl) * 32 + c] * wt2 +
                  OUT[(3 * 32 + pl) * 32 + c] * wt3;
      out[(((size_t)b * 31 + c) * 128 + ii) * 128 + jj] = acc;
    }
  }
}

// =====================================================================
// host launcher
// =====================================================================
extern "C" void kernel_launch(void* const* d_in, const int* in_sizes, int n_in,
                              void* d_out, int out_size, void* d_ws, size_t ws_size,
                              hipStream_t stream) {
  (void)in_sizes; (void)n_in; (void)out_size; (void)ws_size;
  const float* hsi       = (const float*)d_in[0];
  const float* msi       = (const float*)d_in[1];
  const float* enc_hsi_w = (const float*)d_in[2];
  const float* enc_hsi_b = (const float*)d_in[3];
  const float* enc_msi_w = (const float*)d_in[4];
  const float* enc_msi_b = (const float*)d_in[5];
  const float* m_Win     = (const float*)d_in[6];
  const float* m_convw   = (const float*)d_in[7];
  const float* m_convb   = (const float*)d_in[8];
  const float* m_Wxp     = (const float*)d_in[9];
  const float* m_Wdt     = (const float*)d_in[10];
  const float* m_bdt     = (const float*)d_in[11];
  const float* m_Alog    = (const float*)d_in[12];
  const float* m_D       = (const float*)d_in[13];
  const float* m_Wout    = (const float*)d_in[14];
  const float* mlp_W1    = (const float*)d_in[15];
  const float* mlp_b1    = (const float*)d_in[16];
  const float* mlp_W2    = (const float*)d_in[17];
  const float* mlp_b2    = (const float*)d_in[18];
  const float* mlp_W3    = (const float*)d_in[19];
  const float* mlp_b3    = (const float*)d_in[20];
  float* outp = (float*)d_out;

  // ---- bump allocator on d_ws ----
  char* ws = (char*)d_ws;
  size_t off = 0;
  auto alloc = [&](size_t bytes) -> void* {
    void* p = ws + off;
    off += (bytes + 255) & ~(size_t)255;
    return p;
  };
  const size_t BLmax = (size_t)4 * 16384;
  half_t* xh     = (half_t*)alloc(BLmax * 64 * sizeof(half_t));
  float*  xz     = (float*)alloc(BLmax * 256 * sizeof(float));
  half_t* uh     = (half_t*)alloc(BLmax * 128 * sizeof(half_t));
  float*  xdbl   = (float*)alloc(BLmax * 48 * sizeof(float));
  half_t* gh     = (half_t*)alloc(BLmax * 128 * sizeof(half_t));
  float*  E      = (float*)alloc(BLmax * 64 * sizeof(float));   // stage ping
  float*  featb  = (float*)alloc((size_t)4 * 64 * 1024 * sizeof(float));
  float*  guideb = (float*)alloc((size_t)4 * 64 * 16384 * sizeof(float));
  half_t* win_h  = (half_t*)alloc((size_t)4 * 256 * 64 * sizeof(half_t));
  half_t* wxp_h  = (half_t*)alloc((size_t)4 * 48 * 128 * sizeof(half_t));
  half_t* wout_h = (half_t*)alloc((size_t)4 * 64 * 128 * sizeof(half_t));
  half_t* w1h    = (half_t*)alloc((size_t)256 * 160 * sizeof(half_t));
  half_t* w2h    = (half_t*)alloc((size_t)128 * 256 * sizeof(half_t));
  half_t* w3h    = (half_t*)alloc((size_t)32 * 128 * sizeof(half_t));

  auto padcvt = [&](const float* s, half_t* dst, int sr, int sc, int dr, int dc) {
    int n = dr * dc;
    pad_convert_kernel<<<(n + 255) / 256, 256, 0, stream>>>(s, dst, sr, sc, dr, dc);
  };

  // ---- weight prep (f32 -> f16, padded) ----
  padcvt(m_Win, win_h, 4 * 256, 64, 4 * 256, 64);
  for (int i = 0; i < 4; ++i)
    padcvt(m_Wxp + (size_t)i * 36 * 128, wxp_h + (size_t)i * 48 * 128, 36, 128, 48, 128);
  padcvt(m_Wout, wout_h, 4 * 64, 128, 4 * 64, 128);
  padcvt(mlp_W1, w1h, 256, 130, 256, 160);
  padcvt(mlp_W2, w2h, 128, 256, 128, 256);
  padcvt(mlp_W3, w3h, 32, 128, 32, 128);

  // ---- one Mamba block ----
  auto run_mamba = [&](int i, const float* xin, float* xout, int L) {
    int BL = 4 * L;
    convert_f16_kernel<<<(BL * 64 + 255) / 256, 256, 0, stream>>>(xin, xh, (long)BL * 64);
    // xz = x @ Win^T   (M=BL, N=256, K=64)
    gemm_f16_kernel<4><<<dim3(BL / 64, 4), 128, 0, stream>>>(
        xh, win_h + (size_t)i * 256 * 64, xz, BL, 256, 64);
    // u = silu(causal dwconv(xc))
    dwconv_silu_kernel<<<(BL * 128 + 255) / 256, 256, 0, stream>>>(
        xz, m_convw + (size_t)i * 128 * 4, m_convb + (size_t)i * 128, uh, L);
    // xdbl = u @ Wxp^T (M=BL, N=48(pad 36), K=128)
    gemm_f16_kernel<3><<<dim3(BL / 64, 1), 128, 0, stream>>>(
        uh, wxp_h + (size_t)i * 48 * 128, xdbl, BL, 48, 128);
    // fused scan: dt/conv/silu-gate inside, emits gh (f16)
    mamba_scan_kernel<<<4, 128, 0, stream>>>(
        xz, xdbl, m_Alog + (size_t)i * 128 * 16, m_Wdt + (size_t)i * 128 * 4,
        m_bdt + (size_t)i * 128, m_convw + (size_t)i * 128 * 4,
        m_convb + (size_t)i * 128, m_D + (size_t)i * 128, gh, L);
    // out = g @ Wout^T (M=BL, N=64, K=128)
    gemm_f16_kernel<4><<<dim3(BL / 64, 1), 128, 0, stream>>>(
        gh, wout_h + (size_t)i * 64 * 128, xout, BL, 64, 128);
  };

  // ---- feat path (L = 1024) ----
  conv3x3_relu_kernel<<<(4 * 64 * 32 * 32 + 255) / 256, 256, 0, stream>>>(
      hsi, enc_hsi_w, enc_hsi_b, E, 31, 32, 32);
  run_mamba(0, E, E, 1024);
  run_mamba(1, E, featb, 1024);

  // ---- guide path (L = 16384) ----
  conv3x3_relu_kernel<<<(4 * 64 * 128 * 128 + 255) / 256, 256, 0, stream>>>(
      msi, enc_msi_w, enc_msi_b, E, 4, 128, 128);
  run_mamba(2, E, E, 16384);
  run_mamba(3, E, guideb, 16384);

  // ---- fused INR MLP + blend (32 pixels x 4 shifts per block) ----
  const int smem_bytes = 128 * 160 * 2 + 128 * 256 * 2;  // 106496 B of 320KB/WGP
  inr_mlp_kernel<<<(4 * 128 * 128) / 32, 128, smem_bytes, stream>>>(
      featb, guideb, w1h, mlp_b1, w2h, mlp_b2, w3h, mlp_b3, outp);
}
